// PatchEmbeddingR_67851893342540
// MI455X (gfx1250) — matrix operand order repaired and checked by reading, not compiled, not run on hardware
//
#include <hip/hip_runtime.h>
#include <hip/hip_bf16.h>

typedef __attribute__((ext_vector_type(16))) _Float16 v16h;
typedef __attribute__((ext_vector_type(8)))  float    v8f;

#define DM      768      // D_MODEL
#define NBC     1344     // B*C
#define L_IN    1024
#define BM      64       // tokens per block
#define KT      32       // K-step per staged weight tile (48 KB, double buffered)

union FragAB {
    v16h     v;
    uint32_t u[8];
    _Float16 h[16];
};

// ---------------------------------------------------------------------------
// Async global->LDS copy of one 16B chunk (GVS mode: sgpr base + vgpr offset).
// Tracked by ASYNCcnt; no VGPR round trip.
// ---------------------------------------------------------------------------
__device__ __forceinline__ void async_ld_b128(uint32_t lds_off, uint32_t gbyte_off,
                                              const void* gbase) {
    asm volatile("global_load_async_to_lds_b128 %0, %1, %2"
                 :: "v"(lds_off), "v"(gbyte_off), "s"(gbase)
                 : "memory");
}

__device__ __forceinline__ void wait_async0() {
#if __has_builtin(__builtin_amdgcn_s_wait_asynccnt)
    __builtin_amdgcn_s_wait_asynccnt(0);
#else
    asm volatile("s_wait_asynccnt 0x0" ::: "memory");
#endif
}

// Issue async copy of weight tile: cols 0..767, k in [k0, k0+32).
// bstage layout: [col][kk] f16 -> per col 64 bytes = 4 x 16B chunks.
// 3072 chunks / 256 threads = 12 per thread; 8 consecutive threads cover
// two 64B col-rows -> coalesced.
__device__ __forceinline__ void async_tile_copy(uint32_t lds_base,
                                                const _Float16* __restrict__ wt,
                                                int k0, int tid) {
    #pragma unroll
    for (int it = 0; it < 12; ++it) {
        int ch  = tid + it * 256;          // 0..3071
        int col = ch >> 2;
        int q   = ch & 3;
        uint32_t ldso = lds_base + (uint32_t)(col * 64 + q * 16);
        uint32_t goff = (uint32_t)(col * (DM * 2) + k0 * 2 + q * 16);
        async_ld_b128(ldso, goff, (const void*)wt);
    }
}

// ---------------------------------------------------------------------------
// Prep: f32 -> f16 weight conversion.
//   w1t[n*768+k] = w1[k*768+n]   (transposed so K-pairs are contiguous dwords)
//   w2t[n*768+k] = w2[k*768+n]
//   wct[d*64+kk] = conv_w[d][kk&15][kk>>4] for kk<48, else 0   (K padded 48->64)
// ---------------------------------------------------------------------------
__global__ __launch_bounds__(256) void prep_weights(
    const float* __restrict__ conv_w,
    const float* __restrict__ w1,
    const float* __restrict__ w2,
    _Float16* __restrict__ w1t,
    _Float16* __restrict__ w2t,
    _Float16* __restrict__ wct)
{
    int idx = blockIdx.x * 256 + threadIdx.x;
    if (idx < DM * DM) {
        int n = idx / DM;
        int k = idx - n * DM;
        w1t[idx] = (_Float16)w1[k * DM + n];
        w2t[idx] = (_Float16)w2[k * DM + n];
    }
    if (idx < DM * 64) {
        int d  = idx >> 6;
        int kk = idx & 63;
        float v = 0.0f;
        if (kk < 48) {
            int t  = kk >> 4;      // conv tap 0..2
            int ke = kk & 15;      // element within patch
            v = conv_w[d * 48 + ke * 3 + t];
        }
        wct[idx] = (_Float16)v;
    }
}

// ---------------------------------------------------------------------------
// One 64x768 = A(64x768) @ W(768x768) GEMM phase. Weight K-slices staged in
// LDS by async DMA, double buffered; copy of tile k+1 overlaps the 24 WMMAs
// of tile k. Full 64x768 output lives in registers: c[6][4] (192 VGPRs).
// ---------------------------------------------------------------------------
__device__ __forceinline__ void mlp_gemm(
    const _Float16* __restrict__ aSrc,    // LDS A rows [64][768]
    const _Float16* __restrict__ wt,      // global weights [n][k] f16, row=768
    const float*    __restrict__ bias,
    _Float16*       bstage,               // LDS staging: 2 x [768][32] f16
    _Float16*       h1out,                // relu path: f16 output to LDS
    float*          gout,                 // final path: f32 output to global
    const _Float16* resid,                // final path: residual (embh)
    size_t tok0, bool relu)
{
    const int tid   = threadIdx.x;
    const int lane  = tid & 31;
    const int wave  = tid >> 5;
    const int half  = (lane >> 4) & 1;
    const int ln    = lane & 15;
    const int waveM = wave & 3;
    const int waveN = wave >> 2;
    const int mrow  = waveM * 16 + ln;

    const uint32_t bbase0 = (uint32_t)(uintptr_t)bstage;           // LDS offsets
    const uint32_t bbase1 = (uint32_t)(uintptr_t)(bstage + DM * KT);

    v8f c[6][4];
    #pragma unroll
    for (int nt = 0; nt < 6; ++nt)
        #pragma unroll
        for (int s = 0; s < 4; ++s) { v8f z = {}; c[nt][s] = z; }

    // prologue: stage tile 0 into buffer 0
    async_tile_copy(bbase0, wt, 0, tid);
    wait_async0();
    __syncthreads();

    for (int kt = 0; kt < DM / KT; ++kt) {
        const int buf = kt & 1;
        if (kt + 1 < DM / KT)              // overlap: DMA next tile into free buf
            async_tile_copy(buf ? bbase0 : bbase1, wt, (kt + 1) * KT, tid);

        const uint32_t* bs = (const uint32_t*)(bstage + buf * (DM * KT));
        const int ka = kt * KT;

        FragAB a;                          // loaded once, reused by 24 WMMAs
        #pragma unroll
        for (int v = 0; v < 8; ++v) {
            int lk = ((v >> 2) << 4) | (half << 3) | ((v & 3) << 1);
            a.u[v] = *(const uint32_t*)(aSrc + mrow * DM + ka + lk);
        }
        #pragma unroll
        for (int nt = 0; nt < 6; ++nt) {
            #pragma unroll
            for (int s = 0; s < 4; ++s) {
                const int col = nt * 128 + waveN * 64 + s * 16 + ln;
                FragAB bf;                 // two runs of 4 contiguous dwords -> ds_load_b128 x2
                #pragma unroll
                for (int j = 0; j < 8; ++j)
                    bf.u[j] = bs[col * 16 + ((j >> 2) << 3) + half * 4 + (j & 3)];
                c[nt][s] = __builtin_amdgcn_wmma_f32_16x16x32_f16(
                    false, a.v, false, bf.v, (short)0, c[nt][s], false, false);
            }
        }
        wait_async0();     // next tile fully in LDS (wave-local ASYNCcnt)
        __syncthreads();   // all waves done reading buf; writes visible
    }

    // epilogue (C/D layout: lane = N col, VGPR r -> M = 8*half + r)
    #pragma unroll
    for (int nt = 0; nt < 6; ++nt) {
        #pragma unroll
        for (int s = 0; s < 4; ++s) {
            const int col  = nt * 128 + waveN * 64 + s * 16 + ln;
            const float bn = bias[col];
            #pragma unroll
            for (int r = 0; r < 8; ++r) {
                const int row = waveM * 16 + half * 8 + r;
                float vv = c[nt][s][r] + bn;
                if (relu) {
                    h1out[row * DM + col] = (_Float16)(vv > 0.0f ? vv : 0.0f);
                } else {
                    vv += (float)resid[row * DM + col];
                    gout[(tok0 + row) * DM + col] = vv;
                }
            }
        }
    }
}

// ---------------------------------------------------------------------------
// Fused: patchify + circular conv (GEMM, K=64) + MLP (two 768x768 GEMMs)
// + residual. blockIdx.x = bc*2 + half; 64 tokens/block; 8 waves (4M x 2N).
// ---------------------------------------------------------------------------
__global__ __launch_bounds__(256) void fused_patch_mlp(
    const float* __restrict__ x,
    const float* __restrict__ b1,
    const float* __restrict__ b2,
    const _Float16* __restrict__ w1t,
    const _Float16* __restrict__ w2t,
    const _Float16* __restrict__ wct,
    float* __restrict__ out)
{
    __shared__ __align__(16) _Float16 xs[1040];            //  2 KB
    __shared__ __align__(16) _Float16 embh[BM * DM];       // 96 KB
    __shared__ __align__(16) _Float16 h1h [BM * DM];       // 96 KB
    __shared__ __align__(16) _Float16 bstage[2 * DM * KT]; // 2 x 48 KB

    const int tid   = threadIdx.x;
    const int lane  = tid & 31;
    const int wave  = tid >> 5;
    const int half  = (lane >> 4) & 1;
    const int ln    = lane & 15;
    const int waveM = wave & 3;
    const int waveN = wave >> 2;

    const int bc = blockIdx.x >> 1;
    const int p0 = (blockIdx.x & 1) * BM;

    // ---- Phase A: stage x row (edge padded to 1032) as f16 in LDS ----
    const float* xrow = x + (size_t)bc * L_IN;
    for (int i = tid; i < 1032; i += 256) {
        int j = (i < L_IN) ? i : (L_IN - 1);
        xs[i] = (_Float16)xrow[j];
    }
    __syncthreads();

    const uint32_t* wctu = (const uint32_t*)wct;

    const int mrow = waveM * 16 + ln;
    const int p    = p0 + mrow;

    // ---- Phase B: conv GEMM  emb = A_patch(64x64) @ Wc(64x768) ----
    for (int nt = 0; nt < 6; ++nt) {
        const int n0 = nt * 128 + waveN * 64;
        v8f c[4] = {{}, {}, {}, {}};
        #pragma unroll
        for (int ki = 0; ki < 2; ++ki) {
            const int k0 = ki * 32;
            FragAB a;
            #pragma unroll
            for (int v = 0; v < 8; ++v) {
                // 16-bit A 16x32 layout: K = (v/4)*16 + half*8 + (v%4)*2
                int kk = k0 + (((v >> 2) << 4) | (half << 3) | ((v & 3) << 1));
                _Float16 a0 = (_Float16)0.0f, a1 = (_Float16)0.0f;
                if (kk < 48) {
                    int t    = kk >> 4;            // conv tap
                    int q    = (p + t - 1) & 127;  // circular patch index
                    int base = q * 8 + (kk & 15);  // STRIDE=8
                    a0 = xs[base];
                    a1 = xs[base + 1];
                }
                a.h[2 * v]     = a0;
                a.h[2 * v + 1] = a1;
            }
            #pragma unroll
            for (int s = 0; s < 4; ++s) {
                const int col = n0 + s * 16 + ln;
                FragAB bf;
                #pragma unroll
                for (int j = 0; j < 8; ++j) {
                    int kb = k0 + ((j >> 2) << 4) + half * 8 + ((j & 3) << 1);
                    bf.u[j] = wctu[(col << 5) + (kb >> 1)];
                }
                c[s] = __builtin_amdgcn_wmma_f32_16x16x32_f16(
                    false, a.v, false, bf.v, (short)0, c[s], false, false);
            }
        }
        #pragma unroll
        for (int s = 0; s < 4; ++s) {
            const int col = n0 + s * 16 + ln;
            #pragma unroll
            for (int r = 0; r < 8; ++r) {
                int row = waveM * 16 + half * 8 + r;
                embh[row * DM + col] = (_Float16)c[s][r];
            }
        }
    }
    __syncthreads();

    // ---- Phase C: h1 = relu(emb @ w1 + b1) ----
    mlp_gemm(embh, w1t, b1, bstage, h1h, nullptr, nullptr, 0, true);
    __syncthreads();

    // ---- Phase D: out = emb + h1 @ w2 + b2 ----
    mlp_gemm(h1h, w2t, b2, bstage, nullptr, out, embh,
             (size_t)blockIdx.x * BM, false);
}

// ---------------------------------------------------------------------------
extern "C" void kernel_launch(void* const* d_in, const int* in_sizes, int n_in,
                              void* d_out, int out_size, void* d_ws, size_t ws_size,
                              hipStream_t stream) {
    const float* x      = (const float*)d_in[0];   // (64,21,1024)
    const float* conv_w = (const float*)d_in[1];   // (768,16,3)
    const float* w1     = (const float*)d_in[2];   // (768,768)
    const float* b1     = (const float*)d_in[3];   // (768,)
    const float* w2     = (const float*)d_in[4];   // (768,768)
    const float* b2     = (const float*)d_in[5];   // (768,)
    float* out          = (float*)d_out;           // (1344,128,768)

    _Float16* w1t = (_Float16*)d_ws;               // 768*768 f16
    _Float16* w2t = w1t + DM * DM;                 // 768*768 f16
    _Float16* wct = w2t + DM * DM;                 // 768*64  f16 (zero-padded)

    prep_weights<<<(DM * DM + 255) / 256, 256, 0, stream>>>(
        conv_w, w1, w2, w1t, w2t, wct);

    fused_patch_mlp<<<NBC * 2, 256, 0, stream>>>(
        x, b1, b2, w1t, w2t, wct, out);
}
